// Block_59742995087409
// MI455X (gfx1250) — compile-verified, matrix operand order checked by reading, not run on hardware
//
#include <hip/hip_runtime.h>

// ---------------- types / constants ----------------
typedef _Float16 v16h __attribute__((ext_vector_type(16)));
typedef _Float16 h8   __attribute__((ext_vector_type(8)));
typedef float    v8f  __attribute__((ext_vector_type(8)));

#define B_   8
#define T_   2048
#define C_   512
#define H_   8
#define DH_  64
#define MLP_ 2048
#define M_   (B_ * T_)   // 16384 rows

// ---------------- WMMA helpers (gfx1250 layouts) ----------------
// A-frag (16x32 f16): lane L holds row M=L&15; half=(L>>4): elements 0..7 = K(half*8+0..7),
// elements 8..15 = K(16+half*8+0..7). B-frag (32x16) symmetric with lane = N column.
__device__ __forceinline__ v16h mkfrag(h8 lo, h8 hi) {
  return __builtin_shufflevector(lo, hi, 0,1,2,3,4,5,6,7,8,9,10,11,12,13,14,15);
}
// p points at the row's 32-half K chunk; hl = lane>>4
__device__ __forceinline__ v16h load_frag(const _Float16* p, int hl) {
  h8 lo = *(const h8*)(p + hl * 8);
  h8 hi = *(const h8*)(p + 16 + hl * 8);
  return mkfrag(lo, hi);
}
__device__ __forceinline__ v8f wmma_f16(v16h a, v16h b, v8f c) {
  return __builtin_amdgcn_wmma_f32_16x16x32_f16(false, a, false, b, (short)0, c, false, false);
}

// ---------------- f32 -> f16 convert ----------------
__global__ void cvt_f16_kernel(const float* __restrict__ s, _Float16* __restrict__ d, int n) {
  int i = blockIdx.x * 256 + threadIdx.x;
  if (i < n) d[i] = (_Float16)s[i];
}

// ---------------- AdaLN modulation: mod = silu(c) @ ada_w^T + ada_b  [B, 6C] ----------------
__global__ void ada_kernel(const float* __restrict__ c, const float* __restrict__ ada_w,
                           const float* __restrict__ ada_b, float* __restrict__ mod) {
  int idx = blockIdx.x * 256 + threadIdx.x;            // < B_*6*C_ = 24576
  int b = idx / (6 * C_), n = idx % (6 * C_);
  const float* cr = c + (size_t)b * C_;
  const float* wr = ada_w + (size_t)n * C_;
  float acc = ada_b[n];
  for (int k = 0; k < C_; ++k) {
    float v = cr[k];
    acc += (v * (1.f / (1.f + __expf(-v)))) * wr[k];
  }
  mod[idx] = acc;
}

// ---------------- LN + AdaLN modulation -> f16 ----------------
__global__ __launch_bounds__(256)
void ln_mod_kernel(const float* __restrict__ X, const float* __restrict__ w,
                   const float* __restrict__ bb, const float* __restrict__ mod,
                   int off_sc, int off_sh, _Float16* __restrict__ Hout) {
  int row = blockIdx.x;                 // 0..M_-1
  int bidx = row / T_;
  const float* xr = X + (size_t)row * C_;
  float s = 0.f, ss = 0.f;
  #pragma unroll
  for (int cc = threadIdx.x; cc < C_; cc += 256) { float v = xr[cc]; s += v; ss += v * v; }
  #pragma unroll
  for (int m = 16; m; m >>= 1) { s += __shfl_xor(s, m, 32); ss += __shfl_xor(ss, m, 32); }
  __shared__ float rs[8], rss[8];
  int wid = threadIdx.x >> 5, lane = threadIdx.x & 31;
  if (lane == 0) { rs[wid] = s; rss[wid] = ss; }
  __syncthreads();
  if (threadIdx.x == 0) {
    float a = 0.f, q = 0.f;
    #pragma unroll
    for (int i = 0; i < 8; ++i) { a += rs[i]; q += rss[i]; }
    rs[0] = a; rss[0] = q;
  }
  __syncthreads();
  float mean = rs[0] * (1.f / C_);
  float var  = rss[0] * (1.f / C_) - mean * mean;
  float rstd = rsqrtf(var + 1e-5f);
  const float* mrow = mod + (size_t)bidx * (6 * C_);
  #pragma unroll
  for (int cc = threadIdx.x; cc < C_; cc += 256) {
    float v  = (xr[cc] - mean) * rstd * w[cc] + bb[cc];
    float sc = mrow[off_sc + cc], sh = mrow[off_sh + cc];
    Hout[(size_t)row * C_ + cc] = (_Float16)(v * (1.f + sc) + sh);
  }
}

// ---------------- Vt[b][h][d][t] = qkv[b,t, 2C + h*DH + d] ----------------
__global__ void vt_kernel(const _Float16* __restrict__ QKV, _Float16* __restrict__ Vt) {
  int idx = blockIdx.x * 256 + threadIdx.x;   // B*H*DH*T = 8388608
  int t = idx & (T_ - 1);
  int r = idx >> 11;
  int d = r & (DH_ - 1);
  int bh = r >> 6;
  int b = bh >> 3, h = bh & 7;
  Vt[idx] = QKV[((size_t)(b * T_ + t)) * (3 * C_) + 2 * C_ + h * DH_ + d];
}

// ---------------- generic WMMA GEMM: out = act(A[MxK] * W[NxK]^T + bias) ----------------
// MODE 0: store f16 ; MODE 1: GELU(tanh) store f16 ; MODE 2: f32 resid + gate*(acc+bias)
// Double-buffered LDS pipeline: stage tile i+1 before computing tile i.
template<int MODE>
__global__ __launch_bounds__(256)
void gemm_wmma(const _Float16* __restrict__ A, const _Float16* __restrict__ W,
               const float* __restrict__ bias,
               _Float16* __restrict__ outh, float* __restrict__ outf,
               const float* __restrict__ resid, const float* __restrict__ gate,
               int gate_ld, int N, int K) {
  __shared__ _Float16 As[2][128][40];   // 32 K-halves + 8 pad, ping-pong
  __shared__ _Float16 Bs[2][128][40];   // rows indexed by N
  const int m0 = blockIdx.y * 128;
  const int n0 = blockIdx.x * 128;
  const int tid = threadIdx.x;
  const int wave = tid >> 5, lane = tid & 31;
  const int wm = wave >> 1, wn = wave & 1;     // 4x2 wave grid -> 32x64 per wave
  const int hl = lane >> 4, ln16 = lane & 15;

  v8f acc[2][4];
  const v8f vz = {};
  #pragma unroll
  for (int i = 0; i < 2; ++i)
    #pragma unroll
    for (int j = 0; j < 4; ++j) acc[i][j] = vz;

  const int rr = tid >> 1;             // staging row 0..127
  const int ch = (tid & 1) * 16;       // staging col (halves)
  const _Float16* ga = A + (size_t)(m0 + rr) * K + ch;
  const _Float16* gb = W + (size_t)(n0 + rr) * K + ch;

  auto stage = [&](int k0, int buf) {
    const h8* pa = (const h8*)(ga + k0);
    *(h8*)&As[buf][rr][ch]     = pa[0];
    *(h8*)&As[buf][rr][ch + 8] = pa[1];
    const h8* pb = (const h8*)(gb + k0);
    *(h8*)&Bs[buf][rr][ch]     = pb[0];
    *(h8*)&Bs[buf][rr][ch + 8] = pb[1];
  };

  const int nk = K >> 5;
  stage(0, 0);
  for (int i = 0; i < nk; ++i) {
    __syncthreads();
    if (i + 2 < nk) {                       // prime L2 two tiles ahead
      __builtin_prefetch(ga + (i + 2) * 32, 0, 1);
      __builtin_prefetch(gb + (i + 2) * 32, 0, 1);
    }
    if (i + 1 < nk) stage((i + 1) * 32, (i + 1) & 1);
    const int cur = i & 1;
    v16h af[2], bf[4];
    #pragma unroll
    for (int mi = 0; mi < 2; ++mi)
      af[mi] = load_frag(&As[cur][wm * 32 + mi * 16 + ln16][0], hl);
    #pragma unroll
    for (int ni = 0; ni < 4; ++ni)
      bf[ni] = load_frag(&Bs[cur][wn * 64 + ni * 16 + ln16][0], hl);
    #pragma unroll
    for (int mi = 0; mi < 2; ++mi)
      #pragma unroll
      for (int ni = 0; ni < 4; ++ni)
        acc[mi][ni] = wmma_f16(af[mi], bf[ni], acc[mi][ni]);
  }

  #pragma unroll
  for (int mi = 0; mi < 2; ++mi)
    #pragma unroll
    for (int ni = 0; ni < 4; ++ni)
      #pragma unroll
      for (int j = 0; j < 8; ++j) {
        int m = m0 + wm * 32 + mi * 16 + j + hl * 8;   // C/D layout: VGPR j -> row j + 8*half
        int n = n0 + wn * 64 + ni * 16 + ln16;
        float v = acc[mi][ni][j] + bias[n];
        if (MODE == 1) {
          float u = v;
          v = 0.5f * u * (1.f + tanhf(0.7978845608f * (u + 0.044715f * u * u * u)));
        }
        if (MODE == 2) {
          int bbx = m / T_;
          outf[(size_t)m * N + n] =
              resid[(size_t)m * N + n] + gate[(size_t)bbx * gate_ld + n] * v;
        } else {
          outh[(size_t)m * N + n] = (_Float16)v;
        }
      }
}

// ---------------- flash attention ----------------
// 8 waves / 128 Q rows per WG; K and V tiles staged cooperatively into LDS once per WG,
// double-buffered so tile(kt+1) global->LDS overlaps the WMMA work on tile(kt).
// Softmax reductions are batched across the 8 rows per butterfly step so the
// ds_bpermute latency is hidden 8-wide instead of serializing per row.
__global__ __launch_bounds__(256)
void attn_kernel(const _Float16* __restrict__ QKV, const _Float16* __restrict__ Vt,
                 _Float16* __restrict__ O) {
  const int bh = blockIdx.y;
  const int b = bh >> 3, h = bh & 7;
  const int tid = threadIdx.x;
  const int wave = tid >> 5, lane = tid & 31;
  const int hl = lane >> 4, ln16 = lane & 15;
  const int qbase = blockIdx.x * 128 + wave * 16;

  __shared__ _Float16 Ks[2][64][72];   // [t-in-tile][d]  (+8 pad)
  __shared__ _Float16 Vs[2][64][72];   // [d][t-in-tile]
  __shared__ _Float16 Pl[8][16][72];   // per-wave 16x64 P tile

  // stage one 64-wide KV tile (K rows t, V rows d), 256 threads x 16 halves each
  auto stage = [&](int kt, int buf) {
    int r  = tid >> 2;            // 0..63
    int cc = (tid & 3) * 16;      // 0,16,32,48 halves
    const h8* kp = (const h8*)(QKV + (size_t)(b * T_ + kt * 64 + r) * (3 * C_) + C_ + h * DH_ + cc);
    h8* dk = (h8*)&Ks[buf][r][cc];
    dk[0] = kp[0]; dk[1] = kp[1];
    const h8* vp = (const h8*)(Vt + ((size_t)bh * DH_ + r) * T_ + kt * 64 + cc);
    h8* dv = (h8*)&Vs[buf][r][cc];
    dv[0] = vp[0]; dv[1] = vp[1];
  };

  // Q A-fragments (row = t, K = d); fold 1/sqrt(DH)=0.125 into Q once (exact in f16)
  v16h aQ[2];
  {
    const _Float16* qp = QKV + (size_t)(b * T_ + qbase + ln16) * (3 * C_) + h * DH_;
    aQ[0] = load_frag(qp, hl);
    aQ[1] = load_frag(qp + 32, hl);
    const _Float16 qs = (_Float16)0.125f;
    aQ[0] *= qs;
    aQ[1] *= qs;
  }

  float mrow[8], lrow[8];
  v8f of[4];
  const v8f vz = {};
  #pragma unroll
  for (int j = 0; j < 8; ++j) { mrow[j] = -3.0e38f; lrow[j] = 0.f; }
  #pragma unroll
  for (int f = 0; f < 4; ++f) of[f] = vz;

  const int nkt = T_ / 64;
  stage(0, 0);
  for (int kt = 0; kt < nkt; ++kt) {
    __syncthreads();                         // stage(kt) visible; prev reads of buf done
    if (kt + 1 < nkt) stage(kt + 1, (kt + 1) & 1);
    const int cur = kt & 1;

    // S = Q * K^T  (B-frag column = key row t, contraction = d), from LDS
    v8f sf[4];
    #pragma unroll
    for (int f = 0; f < 4; ++f) sf[f] = vz;
    #pragma unroll
    for (int kk = 0; kk < 2; ++kk)
      #pragma unroll
      for (int nf = 0; nf < 4; ++nf) {
        v16h bK = load_frag(&Ks[cur][nf * 16 + ln16][kk * 32], hl);
        sf[nf] = wmma_f16(aQ[kk], bK, sf[nf]);
      }

    // ---- online softmax, batched 8-row reductions ----
    float tm[8], sc8[8], rs8[8];
    #pragma unroll
    for (int j = 0; j < 8; ++j)
      tm[j] = fmaxf(fmaxf(sf[0][j], sf[1][j]), fmaxf(sf[2][j], sf[3][j]));
    #pragma unroll
    for (int mask = 8; mask; mask >>= 1)
      #pragma unroll
      for (int j = 0; j < 8; ++j)
        tm[j] = fmaxf(tm[j], __shfl_xor(tm[j], mask, 32));
    #pragma unroll
    for (int j = 0; j < 8; ++j) {
      float mo = mrow[j];
      float mn = fmaxf(mo, tm[j]);
      sc8[j] = __expf(mo - mn);
      mrow[j] = mn;
      float r = 0.f;
      #pragma unroll
      for (int f = 0; f < 4; ++f) {
        float p = __expf(sf[f][j] - mn);
        sf[f][j] = p;
        r += p;
      }
      rs8[j] = r;
    }
    #pragma unroll
    for (int mask = 8; mask; mask >>= 1)
      #pragma unroll
      for (int j = 0; j < 8; ++j)
        rs8[j] += __shfl_xor(rs8[j], mask, 32);
    #pragma unroll
    for (int j = 0; j < 8; ++j) {
      lrow[j] = lrow[j] * sc8[j] + rs8[j];
      #pragma unroll
      for (int f = 0; f < 4; ++f) of[f][j] *= sc8[j];
      // scatter P into LDS (C-layout -> A-layout exchange, per-wave region)
      #pragma unroll
      for (int f = 0; f < 4; ++f)
        Pl[wave][j + hl * 8][f * 16 + ln16] = (_Float16)sf[f][j];
    }
    asm volatile("s_wait_dscnt 0" ::: "memory");   // per-wave LDS RAW fence

    // O += P * V  (A-frags of P from LDS, B-frags of V from staged Vs)
    v16h aP[2];
    aP[0] = load_frag(&Pl[wave][ln16][0], hl);
    aP[1] = load_frag(&Pl[wave][ln16][32], hl);
    #pragma unroll
    for (int kk = 0; kk < 2; ++kk)
      #pragma unroll
      for (int df = 0; df < 4; ++df) {
        v16h bV = load_frag(&Vs[cur][df * 16 + ln16][kk * 32], hl);
        of[df] = wmma_f16(aP[kk], bV, of[df]);
      }
  }

  #pragma unroll
  for (int df = 0; df < 4; ++df)
    #pragma unroll
    for (int j = 0; j < 8; ++j) {
      int t = qbase + j + hl * 8;
      int d = df * 16 + ln16;
      O[((size_t)(b * T_ + t)) * C_ + h * DH_ + d] = (_Float16)(of[df][j] / lrow[j]);
    }
}

// ---------------- host-side orchestration ----------------
extern "C" void kernel_launch(void* const* d_in, const int* in_sizes, int n_in,
                              void* d_out, int out_size, void* d_ws, size_t ws_size,
                              hipStream_t stream) {
  (void)in_sizes; (void)n_in; (void)out_size; (void)ws_size;
  const float* x      = (const float*)d_in[0];
  const float* c      = (const float*)d_in[1];
  const float* qkv_w  = (const float*)d_in[2];
  const float* qkv_b  = (const float*)d_in[3];
  const float* proj_w = (const float*)d_in[4];
  const float* proj_b = (const float*)d_in[5];
  const float* ada_w  = (const float*)d_in[6];
  const float* ada_b  = (const float*)d_in[7];
  const float* fc1_w  = (const float*)d_in[8];
  const float* fc1_b  = (const float*)d_in[9];
  const float* fc2_w  = (const float*)d_in[10];
  const float* fc2_b  = (const float*)d_in[11];
  const float* ln1_w  = (const float*)d_in[12];
  const float* ln1_b  = (const float*)d_in[13];
  const float* ln2_w  = (const float*)d_in[14];
  const float* ln2_b  = (const float*)d_in[15];
  float* out = (float*)d_out;

  // workspace layout (~141 MB); all chunks are multiples of 256 B
  char* ws = (char*)d_ws;
  size_t off = 0;
  auto alloc = [&](size_t bytes) { void* p = ws + off; off += (bytes + 255) & ~(size_t)255; return p; };
  _Float16* wqkv = (_Float16*)alloc((size_t)3 * C_ * C_ * 2);
  _Float16* wproj= (_Float16*)alloc((size_t)C_ * C_ * 2);
  _Float16* wfc1 = (_Float16*)alloc((size_t)MLP_ * C_ * 2);
  _Float16* wfc2 = (_Float16*)alloc((size_t)C_ * MLP_ * 2);
  float*    mod  = (float*)alloc((size_t)B_ * 6 * C_ * 4);
  _Float16* hbuf = (_Float16*)alloc((size_t)M_ * C_ * 2);
  _Float16* qkvb = (_Float16*)alloc((size_t)M_ * 3 * C_ * 2);
  _Float16* vtb  = (_Float16*)alloc((size_t)B_ * H_ * DH_ * T_ * 2);
  _Float16* obuf = (_Float16*)alloc((size_t)M_ * C_ * 2);
  float*    x1   = (float*)alloc((size_t)M_ * C_ * 4);
  _Float16* m1   = qkvb;   // MLP hidden (67 MB) overlays dead qkv (50 MB) + Vt (17 MB)

  // weight conversion f32 -> f16
  auto cvt = [&](const float* s, _Float16* d, int n) {
    cvt_f16_kernel<<<(n + 255) / 256, 256, 0, stream>>>(s, d, n);
  };
  cvt(qkv_w, wqkv, 3 * C_ * C_);
  cvt(proj_w, wproj, C_ * C_);
  cvt(fc1_w, wfc1, MLP_ * C_);
  cvt(fc2_w, wfc2, C_ * MLP_);

  // AdaLN modulation
  ada_kernel<<<(B_ * 6 * C_) / 256, 256, 0, stream>>>(c, ada_w, ada_b, mod);

  // LN1 + modulate (sh_msa @ 0, sc_msa @ C)
  ln_mod_kernel<<<M_, 256, 0, stream>>>(x, ln1_w, ln1_b, mod, C_, 0, hbuf);

  // qkv = h @ qkv_w^T + b  -> f16 [M, 3C]
  gemm_wmma<0><<<dim3((3 * C_) / 128, M_ / 128), 256, 0, stream>>>(
      hbuf, wqkv, qkv_b, qkvb, nullptr, nullptr, nullptr, 0, 3 * C_, C_);

  // transpose V
  vt_kernel<<<(B_ * H_ * DH_ * T_) / 256, 256, 0, stream>>>(qkvb, vtb);

  // flash attention -> obuf f16 [M, C]
  attn_kernel<<<dim3(T_ / 128, B_ * H_), 256, 0, stream>>>(qkvb, vtb, obuf);

  // x1 = x + g_msa * (o @ proj_w^T + b)   (g_msa @ 2C)
  gemm_wmma<2><<<dim3(C_ / 128, M_ / 128), 256, 0, stream>>>(
      obuf, wproj, proj_b, nullptr, x1, x, mod + 2 * C_, 6 * C_, C_, C_);

  // LN2 + modulate (sh_mlp @ 3C, sc_mlp @ 4C)
  ln_mod_kernel<<<M_, 256, 0, stream>>>(x1, ln2_w, ln2_b, mod, 4 * C_, 3 * C_, hbuf);

  // m1 = gelu(h2 @ fc1_w^T + b) -> f16 [M, MLP]
  gemm_wmma<1><<<dim3(MLP_ / 128, M_ / 128), 256, 0, stream>>>(
      hbuf, wfc1, fc1_b, m1, nullptr, nullptr, nullptr, 0, MLP_, C_);

  // out = x1 + g_mlp * (m1 @ fc2_w^T + b)   (g_mlp @ 5C)
  gemm_wmma<2><<<dim3(C_ / 128, M_ / 128), 256, 0, stream>>>(
      m1, wfc2, fc2_b, nullptr, out, x1, mod + 5 * C_, 6 * C_, C_, MLP_);
}
